// SparseDiffDMC_90426241450488
// MI455X (gfx1250) — compile-verified, matrix operand order checked by reading, not run on hardware
//
#include <hip/hip_runtime.h>
#include <hip/hip_bf16.h>

#define RESMAX 256
#define GRID3 (RESMAX * RESMAX * RESMAX)

typedef __attribute__((ext_vector_type(16))) _Float16 v16h;
typedef __attribute__((ext_vector_type(8)))  float    v8f;

// CUBE_CORNERS order: index = x + 2y + 4z
__constant__ int d_CCX[8] = {0,1,0,1,0,1,0,1};
__constant__ int d_CCY[8] = {0,0,1,1,0,0,1,1};
__constant__ int d_CCZ[8] = {0,0,0,0,1,1,1,1};
// CUBE_EDGES (local corner pairs, original order)
__constant__ int d_CE0[12] = {0,1,4,0,2,3,6,2,2,3,7,6};
__constant__ int d_CE1[12] = {1,5,5,4,3,7,7,6,0,1,5,4};

__global__ void k_fill(int* p, int n, int v) {
  int i = blockIdx.x * blockDim.x + threadIdx.x;
  if (i < n) p[i] = v;
}

// Scatter corner lattice keys (duplicate writes carry identical data) and build voxel grid map.
__global__ void k_scatter(const int* __restrict__ vc, const int* __restrict__ cidx,
                          int N, int* __restrict__ ckey, int* __restrict__ grid) {
  int t = blockIdx.x * blockDim.x + threadIdx.x;
  if (t >= N * 8) return;
  int v = t >> 3, j = t & 7;
  int x = vc[v*3+0], y = vc[v*3+1], z = vc[v*3+2];
  int cx = x + d_CCX[j], cy = y + d_CCY[j], cz = z + d_CCZ[j];
  ckey[cidx[t]] = (cx << 20) | (cy << 10) | cz;
  if (j == 0) grid[(x * 256 + y) * 256 + z] = v;
}

__global__ void k_wpos(const int* __restrict__ ckey, const float* __restrict__ deform,
                       const int* __restrict__ resp, int M, float* __restrict__ wpos) {
  int m = blockIdx.x * blockDim.x + threadIdx.x;
  if (m >= M) return;
  float scale = 2.0f / (float)resp[0];
  int key = ckey[m];
  float cx = (float)((key >> 20) & 1023);
  float cy = (float)((key >> 10) & 1023);
  float cz = (float)(key & 1023);
  wpos[m*3+0] = (cx + 0.5f) * scale - 1.0f + deform[m*3+0];
  wpos[m*3+1] = (cy + 0.5f) * scale - 1.0f + deform[m*3+1];
  wpos[m*3+2] = (cz + 0.5f) * scale - 1.0f + deform[m*3+2];
}

__device__ __forceinline__ int surf_flag_of(const int* cidx, const float* sdf, int v) {
  int nocc = 0;
#pragma unroll
  for (int j = 0; j < 8; ++j) nocc += (sdf[cidx[v*8+j]] < 0.0f) ? 1 : 0;
  return (nocc > 0 && nocc < 8) ? 1 : 0;
}

__global__ void k_surf_count(const int* __restrict__ cidx, const float* __restrict__ sdf,
                             int N, int* __restrict__ bcnt) {
  __shared__ int sh[256];
  int t = threadIdx.x;
  int v = blockIdx.x * 256 + t;
  int flag = (v < N) ? surf_flag_of(cidx, sdf, v) : 0;
  sh[t] = flag; __syncthreads();
  for (int off = 128; off > 0; off >>= 1) { if (t < off) sh[t] += sh[t + off]; __syncthreads(); }
  if (t == 0) bcnt[blockIdx.x] = sh[0];
}

__global__ void k_scan_serial(const int* cnt, int* off, int n, int* total) {
  if (blockIdx.x == 0 && threadIdx.x == 0) {
    int acc = 0;
    for (int i = 0; i < n; ++i) { off[i] = acc; acc += cnt[i]; }
    *total = acc;
  }
}

__global__ void k_surf_scatter(const int* __restrict__ cidx, const float* __restrict__ sdf,
                               int N, const int* __restrict__ boff,
                               int* __restrict__ surf_rank, int* __restrict__ surf_list) {
  __shared__ int sh[256];
  int t = threadIdx.x;
  int v = blockIdx.x * 256 + t;
  int flag = (v < N) ? surf_flag_of(cidx, sdf, v) : 0;
  sh[t] = flag; __syncthreads();
  for (int o = 1; o < 256; o <<= 1) {
    int x = (t >= o) ? sh[t - o] : 0; __syncthreads();
    sh[t] += x; __syncthreads();
  }
  int excl = sh[t] - flag;
  if (v < N) {
    int rank = boff[blockIdx.x] + excl;
    surf_rank[v] = flag ? rank : -1;
    if (flag) surf_list[rank] = v;
  }
}

// vd: 16 surface voxels per wave; 12-edge weighted reduction via V_WMMA_F32_16X16X32_F16
// with B = all-ones (wave-uniform): D[m][*] = sum_k A[m][k].
// A-layout (16-bit A 16x32): lanes 0-15 hold K=0..7, lanes 16-31 hold K=8..15 of row (lane&15),
// so lane l computes products for edges (l<16 ? 0..7 : 8..11) of voxel (l&15).
__global__ __launch_bounds__(32)
void k_vd_wmma(const int* __restrict__ surf_list, const int* __restrict__ cidx,
               const float* __restrict__ sdf, const float* __restrict__ wpos,
               const float* __restrict__ beta, const float* __restrict__ alpha,
               const int* __restrict__ totals, float* __restrict__ out, int out_size) {
  const int S = totals[0];
  const int base = blockIdx.x * 16;
  if (base >= S) return;                 // wave-uniform exit: EXEC stays all-ones past here
  const int lane = threadIdx.x;          // 0..31
  const int row  = lane & 15;
  const int rank = base + row;
  const int rc   = (rank < S) ? rank : (S - 1);  // clamp, no divergence
  const int v    = surf_list[rc];

  int ci[8]; float sv[8]; float as[8];
#pragma unroll
  for (int j = 0; j < 8; ++j) {
    ci[j] = cidx[v*8+j];
    sv[j] = sdf[ci[j]];
    as[j] = tanhf(alpha[v*8+j]) * 0.99f + 1.0f;
  }

  union H16 { v16h v; _Float16 h[16]; };
  H16 ax, ay, az, aw, bOnes;
#pragma unroll
  for (int i = 0; i < 16; ++i) {
    ax.h[i] = (_Float16)0.0f; ay.h[i] = (_Float16)0.0f;
    az.h[i] = (_Float16)0.0f; aw.h[i] = (_Float16)0.0f;
    bOnes.h[i] = (_Float16)1.0f;
  }

  const int slot0 = (lane < 16) ? 0 : 8;
#pragma unroll
  for (int h = 0; h < 8; ++h) {
    int slot = slot0 + h;
    if (slot < 12) {
      int c0 = d_CE0[slot], c1 = d_CE1[slot];
      float sA = sv[c0], sB = sv[c1];
      int crossing = ((sA < 0.0f) != (sB < 0.0f)) ? 1 : 0;
      float bs = tanhf(beta[v*12+slot]) * 0.99f + 1.0f;
      float w  = crossing ? (bs * 0.5f * (as[c0] + as[c1])) : 0.0f;
      float d  = sA - sB;
      float tt = (d != 0.0f) ? (sA / d) : 0.0f;   // crossing => tt in [0,1]
      int a = ci[c0], b = ci[c1];
      float x0 = wpos[a*3+0], y0 = wpos[a*3+1], z0 = wpos[a*3+2];
      float x1 = wpos[b*3+0], y1 = wpos[b*3+1], z1 = wpos[b*3+2];
      float zx = x0 + tt * (x1 - x0);
      float zy = y0 + tt * (y1 - y0);
      float zz = z0 + tt * (z1 - z0);
      ax.h[h] = (_Float16)(w * zx);
      ay.h[h] = (_Float16)(w * zy);
      az.h[h] = (_Float16)(w * zz);
      aw.h[h] = (_Float16)w;
    }
  }

  v8f zeroC = {};
  union F8 { v8f v; float f[8]; };
  F8 dx, dy, dz, dw;
  dx.v = __builtin_amdgcn_wmma_f32_16x16x32_f16(false, ax.v, false, bOnes.v, (short)0, zeroC, false, false);
  dy.v = __builtin_amdgcn_wmma_f32_16x16x32_f16(false, ay.v, false, bOnes.v, (short)0, zeroC, false, false);
  dz.v = __builtin_amdgcn_wmma_f32_16x16x32_f16(false, az.v, false, bOnes.v, (short)0, zeroC, false, false);
  dw.v = __builtin_amdgcn_wmma_f32_16x16x32_f16(false, aw.v, false, bOnes.v, (short)0, zeroC, false, false);

  // C/D layout: row m -> lane 0 (m<8) or lane 16 (m>=8), VGPR m&7 (column 0 == any column here).
  __shared__ float red[2][8][4];
  if (lane == 0 || lane == 16) {
    const int col = lane >> 4;
#pragma unroll
    for (int r2 = 0; r2 < 8; ++r2) {
      red[col][r2][0] = dx.f[r2];
      red[col][r2][1] = dy.f[r2];
      red[col][r2][2] = dz.f[r2];
      red[col][r2][3] = dw.f[r2];
    }
  }
  __syncthreads();
  if (lane < 16 && rank < S) {
    const int col = row >> 3, r2 = row & 7;
    float wsum = red[col][r2][3];
    float inv = (wsum != 0.0f) ? (1.0f / wsum) : 0.0f;
    if (rank * 3 + 3 <= out_size) {
      out[rank*3+0] = red[col][r2][0] * inv;
      out[rank*3+1] = red[col][r2][1] * inv;
      out[rank*3+2] = red[col][r2][2] * inv;
    }
  }
}

// Candidate edge t = 3*m + r, r: 0->z, 1->y, 2->x (matches sorted-key order of the reference).
// Active == crossing && all 4 adjacent cubes are surface voxels (== counts==4).
__device__ __forceinline__ bool edge_eval(int t, int res, const int* __restrict__ ckey,
                                          const int* __restrict__ grid,
                                          const int* __restrict__ surf_rank,
                                          const int* __restrict__ cidx,
                                          const float* __restrict__ sdf,
                                          int rnk[4], bool* flip) {
  int m = t / 3, r = t - 3 * m;
  int key = ckey[m];
  int px = (key >> 20) & 1023, py = (key >> 10) & 1023, pz = key & 1023;
  int axis = (r == 0) ? 2 : ((r == 1) ? 1 : 0);
  int v0 = -1;
  bool ok = true;
  for (int uv = 0; uv < 4; ++uv) {
    int u = uv & 1, w = uv >> 1;
    int cx = px, cy = py, cz = pz;
    if (axis == 2)      { cx -= u; cy -= w; }   // z-edge: cubes (px-u, py-w, pz)
    else if (axis == 1) { cx -= u; cz -= w; }   // y-edge: cubes (px-u, py, pz-w)
    else                { cy -= u; cz -= w; }   // x-edge: cubes (px, py-u, pz-w)
    bool in = (cx >= 0) && (cy >= 0) && (cz >= 0) && (cx < res) && (cy < res) && (cz < res);
    int vv = -1;
    if (in) vv = grid[(cx * 256 + cy) * 256 + cz];
    int sr = (vv >= 0) ? surf_rank[vv] : -1;
    if (sr < 0) ok = false;
    rnk[uv] = sr;
    if (uv == 0) v0 = vv;
  }
  if (!ok) return false;
  int jq = (axis == 2) ? 4 : ((axis == 1) ? 2 : 1);   // q = p + e_axis local corner of cube uv=0
  int m1 = cidx[v0 * 8 + jq];
  float s0 = sdf[m], s1 = sdf[m1];
  if ((s0 < 0.0f) == (s1 < 0.0f)) return false;       // not crossing
  *flip = (s0 > 0.0f);
  // occurrence order within the key group == ascending surf rank: sort ascending
  int a = rnk[0], b = rnk[1], c = rnk[2], d = rnk[3], tmp;
  if (a > b) { tmp = a; a = b; b = tmp; }
  if (c > d) { tmp = c; c = d; d = tmp; }
  if (a > c) { tmp = a; a = c; c = tmp; }
  if (b > d) { tmp = b; b = d; d = tmp; }
  if (b > c) { tmp = b; b = c; c = tmp; }
  rnk[0] = a; rnk[1] = b; rnk[2] = c; rnk[3] = d;
  return true;
}

__global__ void k_edge_count(int E, const int* __restrict__ resp, const int* __restrict__ ckey,
                             const int* __restrict__ grid, const int* __restrict__ surf_rank,
                             const int* __restrict__ cidx, const float* __restrict__ sdf,
                             int* __restrict__ bcnt) {
  __shared__ int sh[256];
  int t0 = threadIdx.x;
  int t = blockIdx.x * 256 + t0;
  int res = resp[0];
  int rnk[4]; bool flip;
  int flag = 0;
  if (t < E) flag = edge_eval(t, res, ckey, grid, surf_rank, cidx, sdf, rnk, &flip) ? 1 : 0;
  sh[t0] = flag; __syncthreads();
  for (int off = 128; off > 0; off >>= 1) { if (t0 < off) sh[t0] += sh[t0 + off]; __syncthreads(); }
  if (t0 == 0) bcnt[blockIdx.x] = sh[0];
}

__global__ void k_faces(int E, const int* __restrict__ resp, const int* __restrict__ ckey,
                        const int* __restrict__ grid, const int* __restrict__ surf_rank,
                        const int* __restrict__ cidx, const float* __restrict__ sdf,
                        const int* __restrict__ boff, const int* __restrict__ totals,
                        float* __restrict__ out, int out_size) {
  __shared__ int sh[256];
  int t0 = threadIdx.x;
  int t = blockIdx.x * 256 + t0;
  int res = resp[0];
  int rnk[4]; bool flip = false;
  int flag = 0;
  if (t < E) flag = edge_eval(t, res, ckey, grid, surf_rank, cidx, sdf, rnk, &flip) ? 1 : 0;
  sh[t0] = flag; __syncthreads();
  for (int o = 1; o < 256; o <<= 1) {
    int x = (t0 >= o) ? sh[t0 - o] : 0; __syncthreads();
    sh[t0] += x; __syncthreads();
  }
  int excl = sh[t0] - flag;
  if (flag) {
    int pos = boff[blockIdx.x] + excl;
    int S = totals[0];
    int q0, q1, q2, q3;
    if (flip) { q0 = rnk[3]; q1 = rnk[2]; q2 = rnk[1]; q3 = rnk[0]; }
    else      { q0 = rnk[0]; q1 = rnk[1]; q2 = rnk[2]; q3 = rnk[3]; }
    int o6 = 3 * S + pos * 6;
    if (o6 + 6 <= out_size) {
      out[o6+0] = (float)q0; out[o6+1] = (float)q1; out[o6+2] = (float)q2;
      out[o6+3] = (float)q0; out[o6+4] = (float)q2; out[o6+5] = (float)q3;
    }
  }
}

extern "C" void kernel_launch(void* const* d_in, const int* in_sizes, int n_in,
                              void* d_out, int out_size, void* d_ws, size_t ws_size,
                              hipStream_t stream) {
  const int*   vc     = (const int*)d_in[0];    // N*3 int32
  const int*   cidx   = (const int*)d_in[1];    // N*8 int32
  const float* sdf    = (const float*)d_in[2];  // M
  const float* deform = (const float*)d_in[3];  // M*3
  const float* beta   = (const float*)d_in[4];  // N*12
  const float* alpha  = (const float*)d_in[5];  // N*8
  const int*   resp   = (const int*)d_in[7];    // scalar resolution
  (void)n_in; (void)d_in;  // gamma (d_in[6]) unused by the reference too

  const int N  = in_sizes[1] / 8;
  const int M  = in_sizes[2];
  const int E  = 3 * M;
  const int B1 = (N + 255) / 256;
  const int B2 = (E + 255) / 256;

  char* p = (char*)d_ws;
  auto take = [&](size_t b) -> void* {
    void* r = (void*)p;
    p += (b + 255) & ~(size_t)255;
    return r;
  };
  int*   grid      = (int*)take(sizeof(int) * (size_t)GRID3);
  int*   ckey      = (int*)take(sizeof(int) * (size_t)M);
  float* wpos      = (float*)take(sizeof(float) * 3 * (size_t)M);
  int*   surf_rank = (int*)take(sizeof(int) * (size_t)N);
  int*   surf_list = (int*)take(sizeof(int) * (size_t)N);
  int*   bcnt1     = (int*)take(sizeof(int) * (size_t)B1);
  int*   boff1     = (int*)take(sizeof(int) * (size_t)B1);
  int*   bcnt2     = (int*)take(sizeof(int) * (size_t)B2);
  int*   boff2     = (int*)take(sizeof(int) * (size_t)B2);
  int*   totals    = (int*)take(sizeof(int) * 4);
  (void)ws_size;

  float* out = (float*)d_out;

  k_fill<<<(GRID3 + 255) / 256, 256, 0, stream>>>(grid, GRID3, -1);
  k_scatter<<<(N * 8 + 255) / 256, 256, 0, stream>>>(vc, cidx, N, ckey, grid);
  k_wpos<<<(M + 255) / 256, 256, 0, stream>>>(ckey, deform, resp, M, wpos);
  k_surf_count<<<B1, 256, 0, stream>>>(cidx, sdf, N, bcnt1);
  k_scan_serial<<<1, 1, 0, stream>>>(bcnt1, boff1, B1, &totals[0]);
  k_surf_scatter<<<B1, 256, 0, stream>>>(cidx, sdf, N, boff1, surf_rank, surf_list);
  const int tiles = (N + 15) / 16;  // upper bound on S/16; tail waves exit uniformly
  k_vd_wmma<<<tiles, 32, 0, stream>>>(surf_list, cidx, sdf, wpos, beta, alpha, totals, out, out_size);
  k_edge_count<<<B2, 256, 0, stream>>>(E, resp, ckey, grid, surf_rank, cidx, sdf, bcnt2);
  k_scan_serial<<<1, 1, 0, stream>>>(bcnt2, boff2, B2, &totals[1]);
  k_faces<<<B2, 256, 0, stream>>>(E, resp, ckey, grid, surf_rank, cidx, sdf, boff2, totals, out, out_size);
}
